// DeepTransform_58050777973407
// MI455X (gfx1250) — matrix operand order repaired and checked by reading, not compiled, not run on hardware
//
#include <hip/hip_runtime.h>

// ---------------- problem constants ----------------
#define LYR 3
#define Hc 256
#define Fc 256
#define H3 768     // 3*H
#define H2 512     // 2*H
#define Hh 128     // H/2

#define INV_SQRT3 0.5773502691896258f
#define INV_SQRT2 0.7071067811865476f
#define INV_SQRTH 0.0625f            // 1/sqrt(256)
#define SILU_SCALE 1.6666666666666667f

// LDS row stride (floats) for the staged 16x256 edge tile: 260 mod 64 == 4, so the
// 16 rows read by a 16-lane group start in 16 distinct banks (conflict-free b128s).
#define EF_LDS_STRIDE 260

typedef __attribute__((ext_vector_type(16))) _Float16 v16h;
typedef __attribute__((ext_vector_type(8)))  float    v8f;

// type matching the async-LDS builtin's parameter: vector_size(16) int
typedef int b128i __attribute__((vector_size(16)));
typedef __attribute__((address_space(1))) b128i* gptr_b128;
typedef __attribute__((address_space(3))) b128i* lptr_b128;

#if defined(__has_builtin)
#if __has_builtin(__builtin_amdgcn_global_load_async_to_lds_b128)
#define HAS_ASYNC_LDS 1
#endif
#endif

__device__ __forceinline__ float scaled_silu(float x) {
    return x * (1.0f / (1.0f + __expf(-x))) * SILU_SCALE;
}

__device__ __forceinline__ void atomic_add_f32(float* p, float v) {
#if defined(__HIP_PLATFORM_AMD__)
    unsafeAtomicAdd(p, v);   // hardware global_atomic_add_f32
#else
    atomicAdd(p, v);
#endif
}

// Build 16-bit A fragment (16x32 MxK tile) for wave32 WMMA.
// Lane holds row m = lane&15; kh = lane>>4 selects K-halves:
//   a[0..15] <- K = 16*kh + 0..15
__device__ __forceinline__ v16h load_a_frag_f32(const float* ap, int kh) {
    const float4 p0 = *(const float4*)(ap + 16 * kh);
    const float4 p1 = *(const float4*)(ap + 16 * kh + 4);
    const float4 q0 = *(const float4*)(ap + 16 * kh + 8);
    const float4 q1 = *(const float4*)(ap + 16 * kh + 12);
    v16h a;
    a[0]=(_Float16)p0.x;  a[1]=(_Float16)p0.y;  a[2]=(_Float16)p0.z;  a[3]=(_Float16)p0.w;
    a[4]=(_Float16)p1.x;  a[5]=(_Float16)p1.y;  a[6]=(_Float16)p1.z;  a[7]=(_Float16)p1.w;
    a[8]=(_Float16)q0.x;  a[9]=(_Float16)q0.y;  a[10]=(_Float16)q0.z; a[11]=(_Float16)q0.w;
    a[12]=(_Float16)q1.x; a[13]=(_Float16)q1.y; a[14]=(_Float16)q1.z; a[15]=(_Float16)q1.w;
    return a;
}

// ---------------- weight prep: f32 [L,K,N] -> f16 transposed [L,N,K] ----------------
__global__ void k_prep_wt(const float* __restrict__ in, _Float16* __restrict__ out,
                          int total, int K, int N) {
    int idx = blockIdx.x * blockDim.x + threadIdx.x;
    if (idx >= total) return;
    int n = idx % N;
    int k = (idx / N) % K;
    int l = idx / (N * K);
    out[(size_t)l * N * K + (size_t)n * K + k] = (_Float16)in[idx];
}

// ---------------- elementwise helpers ----------------
__global__ void k_copy(const float* __restrict__ s, float* __restrict__ d, size_t n) {
    size_t i = (size_t)blockIdx.x * blockDim.x + threadIdx.x;
    if (i < n) d[i] = s[i];
}
__global__ void k_zero(float* __restrict__ d, size_t n) {
    size_t i = (size_t)blockIdx.x * blockDim.x + threadIdx.x;
    if (i < n) d[i] = 0.0f;
}
__global__ void k_scale_copy(const float* __restrict__ s, float* __restrict__ d,
                             float sc, size_t n) {
    size_t i = (size_t)blockIdx.x * blockDim.x + threadIdx.x;
    if (i < n) d[i] = s[i] * sc;
}

// ---------------- generic WMMA GEMM ----------------
// C[M,Nc] = act( concat(A0,A1)[M,K] @ BT[Nc,K]^T + bias )
// One wave (32 threads) per block computes a 16x64 tile.
// A is f32 row-major (converted to f16 on the fly); BT is f16 [Nc,K] (pre-transposed).
// splitK: K index where A1 begins (multiple of 32). For plain GEMM pass splitK=K, A1=A0.
__global__ void k_gemm_wmma(const float* __restrict__ A0, const float* __restrict__ A1,
                            int splitK, const _Float16* __restrict__ BT,
                            const float* __restrict__ bias, float* __restrict__ C,
                            int Nc, int K, int act) {
    const int lane = threadIdx.x;
    const int r  = lane & 15;
    const int kh = lane >> 4;
    const int row = blockIdx.x * 16 + r;
    const int cb  = blockIdx.y * 64;
    const int lda1 = K - splitK;

    v8f acc[4] = {};
    for (int kb = 0; kb < K; kb += 32) {
        const float* ap = (kb < splitK)
            ? (A0 + (size_t)row * splitK + kb)
            : (A1 + (size_t)row * lda1 + (kb - splitK));
        const v16h a = load_a_frag_f32(ap, kh);
#pragma unroll
        for (int t = 0; t < 4; ++t) {
            const _Float16* bp = BT + (size_t)(cb + 16 * t + r) * K + kb + 16 * kh;
            const v16h b = *(const v16h*)bp;
            acc[t] = __builtin_amdgcn_wmma_f32_16x16x32_f16(
                false, a, false, b, (short)0, acc[t], false, false);
        }
    }
    // C/D layout: lane holds col n=lane&15, rows v + 8*(lane>>4)
    const int rowbase = blockIdx.x * 16 + 8 * kh;
#pragma unroll
    for (int t = 0; t < 4; ++t) {
        const int col = cb + 16 * t + r;
        const float bv = bias ? bias[col] : 0.0f;
#pragma unroll
        for (int v = 0; v < 8; ++v) {
            float val = acc[t][v] + bv;
            if (act) val = scaled_silu(val);
            C[(size_t)(rowbase + v) * Nc + col] = val;
        }
    }
}

// ---------------- fused edge kernel ----------------
// Per block: 16 edges, 256 threads (8 waves). The 16x256 f32 edge_feat tile is staged
// once into LDS (async global->LDS when available), then each wave w computes columns
// [w*96, w*96+96) of rbf = edge_feat @ mp_ew via WMMA over K=256, adds bias,
// multiplies by the gathered x_h[j] row, and scatter-adds:
//   c in [0,H):    vec_new[i][d][c]   += m * vec_old[j][d][c]   * INV_SQRTH
//   c in [H,2H):   vec_new[i][d][c-H] += m * edge_vector[e][d]  * INV_SQRTH
//   c in [2H,3H):  xacc[i][c-2H]      += m
// No [E,3H] intermediate ever touches memory; node state stays L2-resident.
__global__ void k_edge_fused(const float* __restrict__ ef, const _Float16* __restrict__ ewT,
                             const float* __restrict__ eb, const float* __restrict__ x_h,
                             const float* __restrict__ vec_old, const float* __restrict__ evec,
                             const int* __restrict__ eidx, int E,
                             float* __restrict__ vec_new, float* __restrict__ xacc) {
    __shared__ float smem[16 * EF_LDS_STRIDE];

    const int tid  = threadIdx.x;
    const int lane = tid & 31;
    const int wave = tid >> 5;
    const int r  = lane & 15;
    const int kh = lane >> 4;
    const int ebase = blockIdx.x * 16;

    // ---- stage 16 rows x 256 f32 (16 KB) into LDS, padded rows ----
#pragma unroll
    for (int kchunk = 0; kchunk < 4; ++kchunk) {
        const int ci   = tid + kchunk * 256;   // 1024 x 16B chunks
        const int rrow = ci >> 6;              // 64 chunks per row
        const int cc   = (ci & 63) << 2;       // float offset in row
        const float* gsrc = ef + (size_t)(ebase + rrow) * Fc + cc;
        float* ldst = &smem[rrow * EF_LDS_STRIDE + cc];
#if defined(HAS_ASYNC_LDS)
        __builtin_amdgcn_global_load_async_to_lds_b128(
            (gptr_b128)gsrc, (lptr_b128)ldst, 0, 0);
#else
        *(float4*)ldst = *(const float4*)gsrc;
#endif
    }
    // prefetch next tile's edge_feat stream (16 KB) while we compute
    if (ebase + 16 < E)
        __builtin_prefetch((const void*)(ef + (size_t)(ebase + 16) * Fc + (size_t)tid * 16), 0, 0);
#if defined(HAS_ASYNC_LDS)
#if __has_builtin(__builtin_amdgcn_s_wait_asynccnt)
    __builtin_amdgcn_s_wait_asynccnt(0);
#endif
#endif
    __syncthreads();

    // ---- WMMA: rbf tile for this wave's 96 columns ----
    const float* arow = &smem[r * EF_LDS_STRIDE];
    v8f acc[6] = {};
    for (int kb = 0; kb < Fc; kb += 32) {
        const v16h a = load_a_frag_f32(arow + kb, kh);
#pragma unroll
        for (int t = 0; t < 6; ++t) {
            const int col = wave * 96 + 16 * t + r;
            const v16h b = *(const v16h*)(ewT + (size_t)col * Fc + kb + 16 * kh);
            acc[t] = __builtin_amdgcn_wmma_f32_16x16x32_f16(
                false, a, false, b, (short)0, acc[t], false, false);
        }
    }

    // edges handled by this lane's accumulator rows: e = ebase + v + 8*kh
    int js[8], is_[8];
#pragma unroll
    for (int v = 0; v < 8; ++v) {
        const int e = ebase + v + 8 * kh;
        js[v]  = eidx[e];
        is_[v] = eidx[E + e];
    }

#pragma unroll
    for (int t = 0; t < 6; ++t) {
        const int c = wave * 96 + 16 * t + r;
        const float ebv = eb[c];
        const int region = c >> 8;   // uniform per (wave,t): 16-col tiles never straddle
#pragma unroll
        for (int v = 0; v < 8; ++v) {
            const int e = ebase + v + 8 * kh;
            const float rbf = acc[t][v] + ebv;
            const float m = x_h[(size_t)js[v] * H3 + c] * rbf * INV_SQRT3;
            if (region == 0) {
                const size_t jb = (size_t)js[v] * H3 + c;
                const size_t ib = (size_t)is_[v] * H3 + c;
#pragma unroll
                for (int d = 0; d < 3; ++d)
                    atomic_add_f32(&vec_new[ib + d * Hc],
                                   m * vec_old[jb + d * Hc] * INV_SQRTH);
            } else if (region == 1) {
                const size_t ib = (size_t)is_[v] * H3 + (c - Hc);
#pragma unroll
                for (int d = 0; d < 3; ++d)
                    atomic_add_f32(&vec_new[ib + d * Hc],
                                   m * evec[(size_t)e * 3 + d] * INV_SQRTH);
            } else {
                atomic_add_f32(&xacc[(size_t)is_[v] * Hc + (c - H2)], m);
            }
        }
    }
}

// ---------------- vec_dot / vnorm ----------------
// vp is [N,3,2H]; vec1 = vp[..., :H], vec2 = vp[..., H:]
__global__ void k_vdotnorm(const float* __restrict__ vp, float* __restrict__ vdot,
                           float* __restrict__ vnorm, int NH) {
    int idx = blockIdx.x * blockDim.x + threadIdx.x;
    if (idx >= NH) return;
    const int n = idx >> 8;
    const int h = idx & 255;
    const size_t base = (size_t)n * 3 * H2 + h;
    float s = 0.f, s2 = 0.f;
#pragma unroll
    for (int d = 0; d < 3; ++d) {
        const float a = vp[base + (size_t)d * H2];
        const float b = vp[base + (size_t)d * H2 + Hc];
        s  += a * b;
        s2 += b * b;
    }
    vdot[idx]  = s * INV_SQRTH;
    vnorm[idx] = sqrtf(s2 + 1e-8f);
}

// ---------------- final per-layer update ----------------
// x += (xv1 + xv2*vdot)*INV_SQRT2 ; vec += xv3 * vec1
__global__ void k_update(const float* __restrict__ xh, const float* __restrict__ vdot,
                         const float* __restrict__ vp, float* __restrict__ x,
                         float* __restrict__ vec, int NH) {
    int idx = blockIdx.x * blockDim.x + threadIdx.x;
    if (idx >= NH) return;
    const int n = idx >> 8;
    const int h = idx & 255;
    const float* row = xh + (size_t)n * H3;
    const float xv1 = row[h];
    const float xv2 = row[Hc + h];
    const float xv3 = row[H2 + h];
    x[idx] += (xv1 + xv2 * vdot[idx]) * INV_SQRT2;
    const size_t vb = (size_t)n * H3 + h;
    const size_t pb = (size_t)n * 3 * H2 + h;
#pragma unroll
    for (int d = 0; d < 3; ++d)
        vec[vb + (size_t)d * Hc] += xv3 * vp[pb + (size_t)d * H2];
}

// ---------------- host side ----------------
static inline int cdiv(size_t a, int b) { return (int)((a + b - 1) / b); }

extern "C" void kernel_launch(void* const* d_in, const int* in_sizes, int n_in,
                              void* d_out, int out_size, void* d_ws, size_t ws_size,
                              hipStream_t stream) {
    const float* x_in  = (const float*)d_in[0];
    const float* ef    = (const float*)d_in[1];
    const float* evec  = (const float*)d_in[2];
    const int*   eidx  = (const int*)  d_in[3];
    const float* mp_w1 = (const float*)d_in[4];
    const float* mp_b1 = (const float*)d_in[5];
    const float* mp_w2 = (const float*)d_in[6];
    const float* mp_b2 = (const float*)d_in[7];
    const float* mp_ew = (const float*)d_in[8];
    const float* mp_eb = (const float*)d_in[9];
    const float* mu_vw = (const float*)d_in[10];
    const float* mu_w1 = (const float*)d_in[11];
    const float* mu_b1 = (const float*)d_in[12];
    const float* mu_w2 = (const float*)d_in[13];
    const float* mu_b2 = (const float*)d_in[14];

    const int N = in_sizes[0] / Hc;      // 10000
    const int E = in_sizes[3] / 2;       // 160000
    const size_t NH  = (size_t)N * Hc;
    const size_t NV  = (size_t)N * H3;   // N*3*H

    // outputs double as running state
    float* x    = (float*)d_out;         // [N,H]
    float* vecB = (float*)d_out + NH;    // [N,3,H]  (final vec lands here)

    // ---- workspace carve-up ----
    _Float16* wt = (_Float16*)d_ws;
    const size_t SZ_W1T  = (size_t)LYR * Hh * Hc;
    const size_t SZ_W2T  = (size_t)LYR * H3 * Hh;
    const size_t SZ_EWT  = (size_t)LYR * H3 * Fc;
    const size_t SZ_VWT  = (size_t)LYR * H2 * Hc;
    const size_t SZ_MW1T = (size_t)LYR * Hc * H2;
    const size_t SZ_MW2T = (size_t)LYR * H3 * Hc;
    _Float16* w1T  = wt;
    _Float16* w2T  = w1T  + SZ_W1T;
    _Float16* ewT  = w2T  + SZ_W2T;
    _Float16* vwT  = ewT  + SZ_EWT;
    _Float16* mw1T = vwT  + SZ_VWT;
    _Float16* mw2T = mw1T + SZ_MW1T;
    const size_t WT_HALVES = SZ_W1T + SZ_W2T + SZ_EWT + SZ_VWT + SZ_MW1T + SZ_MW2T;

    float* f32b  = (float*)((char*)d_ws + WT_HALVES * sizeof(_Float16));
    float* vecA  = f32b;                 // [N,3,H]
    float* xacc  = vecA  + NV;           // [N,H]
    float* h1    = xacc  + NH;           // [N,H/2]
    float* xh    = h1    + (size_t)N * Hh; // [N,3H] (mp x_h, reused for mu xh)
    float* vp    = xh    + NV;           // [N,3,2H]
    float* vdot  = vp    + (size_t)N * 3 * H2; // [N,H]
    float* vnorm = vdot  + NH;           // [N,H]
    float* hmid  = vnorm + NH;           // [N,H]

    const int TB = 256;

    // ---- weight prep (deterministic, every call) ----
    {
        int t;
        t = LYR * Hc * Hh; k_prep_wt<<<cdiv(t, TB), TB, 0, stream>>>(mp_w1, w1T,  t, Hc, Hh);
        t = LYR * Hh * H3; k_prep_wt<<<cdiv(t, TB), TB, 0, stream>>>(mp_w2, w2T,  t, Hh, H3);
        t = LYR * Fc * H3; k_prep_wt<<<cdiv(t, TB), TB, 0, stream>>>(mp_ew, ewT,  t, Fc, H3);
        t = LYR * Hc * H2; k_prep_wt<<<cdiv(t, TB), TB, 0, stream>>>(mu_vw, vwT,  t, Hc, H2);
        t = LYR * H2 * Hc; k_prep_wt<<<cdiv(t, TB), TB, 0, stream>>>(mu_w1, mw1T, t, H2, Hc);
        t = LYR * Hc * H3; k_prep_wt<<<cdiv(t, TB), TB, 0, stream>>>(mu_w2, mw2T, t, Hc, H3);
    }

    // ---- init running state ----
    k_copy<<<cdiv(NH, TB), TB, 0, stream>>>(x_in, x, NH);
    k_zero<<<cdiv(NV, TB), TB, 0, stream>>>(vecA, NV);

    for (int l = 0; l < LYR; ++l) {
        const float* vec_old = (l & 1) ? vecB : vecA;  // l0:A(zeros) l1:B l2:A
        float*       vec_new = (l & 1) ? vecA : vecB;  // l2 -> vecB = d_out

        // --- MessagePassing node MLP: h1 = silu(x@w1+b1); x_h = h1@w2+b2 ---
        {
            dim3 g1(N / 16, Hh / 64);
            k_gemm_wmma<<<g1, 32, 0, stream>>>(x, x, Hc, w1T + (size_t)l * Hh * Hc,
                                               mp_b1 + (size_t)l * Hh, h1, Hh, Hc, 1);
            dim3 g2(N / 16, H3 / 64);
            k_gemm_wmma<<<g2, 32, 0, stream>>>(h1, h1, Hh, w2T + (size_t)l * H3 * Hh,
                                               mp_b2 + (size_t)l * H3, xh, H3, Hh, 0);
        }

        // --- scatter targets: vec_new = vec_old (copy), xacc = x ---
        k_copy<<<cdiv(NV, TB), TB, 0, stream>>>(vec_old, vec_new, NV);
        k_copy<<<cdiv(NH, TB), TB, 0, stream>>>(x, xacc, NH);

        // --- fused rbf-GEMM + gather + edge math + scatter ---
        k_edge_fused<<<E / 16, 256, 0, stream>>>(
            ef, ewT + (size_t)l * H3 * Fc, mp_eb + (size_t)l * H3,
            xh, vec_old, evec, eidx, E, vec_new, xacc);

        // --- x = (x + dx) * 1/sqrt(2) ---
        k_scale_copy<<<cdiv(NH, TB), TB, 0, stream>>>(xacc, x, INV_SQRT2, NH);

        // --- MessageUpdating ---
        {
            dim3 g3((N * 3) / 16, H2 / 64);  // vp = vec_new[N*3,H] @ mu_vw
            k_gemm_wmma<<<g3, 32, 0, stream>>>(vec_new, vec_new, Hc,
                                               vwT + (size_t)l * H2 * Hc,
                                               nullptr, vp, H2, Hc, 0);
        }
        k_vdotnorm<<<cdiv(NH, TB), TB, 0, stream>>>(vp, vdot, vnorm, (int)NH);
        {
            dim3 g4(N / 16, Hc / 64);        // hmid = silu(concat(x,vnorm)@mu_w1+b1)
            k_gemm_wmma<<<g4, 32, 0, stream>>>(x, vnorm, Hc,
                                               mw1T + (size_t)l * Hc * H2,
                                               mu_b1 + (size_t)l * Hc, hmid, Hc, H2, 1);
            dim3 g5(N / 16, H3 / 64);        // xh = hmid@mu_w2+b2
            k_gemm_wmma<<<g5, 32, 0, stream>>>(hmid, hmid, Hc,
                                               mw2T + (size_t)l * H3 * Hc,
                                               mu_b2 + (size_t)l * H3, xh, H3, Hc, 0);
        }
        k_update<<<cdiv(NH, TB), TB, 0, stream>>>(xh, vdot, vp, x, vec_new, (int)NH);
    }
}
